// DrugGraphEncoder_60052232733141
// MI455X (gfx1250) — matrix operand order repaired and checked by reading, not compiled, not run on hardware
//
#include <hip/hip_runtime.h>

// ---------------- problem constants ----------------
#define N_    131072
#define E_    524288
#define ET_   (E_ + N_)          // edges + self loops
#define INDIM 78
#define HID   64
#define HEADS 4
#define DOUT  256                // HEADS*HID
#define KPAD  96                 // padded K for layer 0 (78 -> 96), layers 1/2 use first 64
#define G_    4096
#define NSLOPE 0.2f

typedef __attribute__((ext_vector_type(16))) __bf16 v16bf;
typedef __attribute__((ext_vector_type(8)))  __bf16 v8bf;
typedef __attribute__((ext_vector_type(8)))  float  v8f;

// ---------------- helpers ----------------
__global__ void set_f32_kernel(float* __restrict__ p, float v, unsigned long long n) {
  unsigned long long i = (unsigned long long)blockIdx.x * blockDim.x + threadIdx.x;
  unsigned long long stride = (unsigned long long)gridDim.x * blockDim.x;
  for (; i < n; i += stride) p[i] = v;
}

__device__ __forceinline__ void atomicMaxF32(float* addr, float val) {
  // monotonic float-max via integer atomics (valid for non-NaN, -inf init)
  if (val >= 0.0f) {
    atomicMax((int*)addr, __float_as_int(val));
  } else {
    atomicMin((unsigned int*)addr, (unsigned int)__float_as_int(val));
  }
}

// ---------------- input prep ----------------
__global__ void convert_x_kernel(const float* __restrict__ x, __bf16* __restrict__ xb) {
  int idx = blockIdx.x * blockDim.x + threadIdx.x;      // over N_*KPAD
  int n = idx / KPAD, k = idx - n * KPAD;
  float v = (k < INDIM) ? x[(size_t)n * INDIM + k] : 0.0f;
  xb[idx] = (__bf16)v;
}

__global__ void prep_w_kernel(const float* __restrict__ W, __bf16* __restrict__ Wt, int din) {
  int idx = blockIdx.x * blockDim.x + threadIdx.x;      // over DOUT*KPAD
  int col = idx / KPAD, k = idx - col * KPAD;
  float v = (k < din) ? W[(size_t)k * DOUT + col] : 0.0f;
  Wt[(size_t)col * KPAD + k] = (__bf16)v;
}

// ---------------- WMMA GEMM: H[N,256] = Xb[N,KPAD] * W[K,256] ----------------
// 1 block = 4 waves; block computes rows [16*bid, 16*bid+16), wave w covers cols [64w, 64w+64)
__global__ __launch_bounds__(128)
void gemm_wmma_kernel(const __bf16* __restrict__ Xb, const __bf16* __restrict__ Wt,
                      float* __restrict__ H, int ksteps) {
  const int wave = threadIdx.x >> 5;
  const int lane = threadIdx.x & 31;
  const int m    = lane & 15;       // row within tile (A), col within tile (B/C)
  const int hi   = lane >> 4;       // lane-half selector
  const int mbase = blockIdx.x * 16;
  const int nbase = wave * 64;

  const __bf16* arow = Xb + (size_t)(mbase + m) * KPAD;

  v8f acc0 = {}, acc1 = {}, acc2 = {}, acc3 = {};

  for (int ks = 0; ks < ksteps; ++ks) {
    const int kbase = ks * 32;
    __builtin_prefetch(arow + kbase + 32, 0, 0);   // global_prefetch_b8 of next K-step

    // A fragment: 16-bit A 16x32 layout; element e <-> K = kbase + 8*hi + (e&7) + (e>=8?16:0)
    v8bf alo = *(const v8bf*)(arow + kbase + 8 * hi);
    v8bf ahi = *(const v8bf*)(arow + kbase + 16 + 8 * hi);
    v16bf afrag;
#pragma unroll
    for (int i = 0; i < 8; ++i) { afrag[i] = alo[i]; afrag[i + 8] = ahi[i]; }

#pragma unroll
    for (int nt = 0; nt < 4; ++nt) {
      // B fragment: lane holds column (nbase+nt*16+m), element e <-> K = kbase + 8*hi + (e&7) + (e>=8?16:0)
      const __bf16* bp = Wt + (size_t)(nbase + nt * 16 + m) * KPAD + kbase + 8 * hi;
      v8bf blo = *(const v8bf*)(bp);
      v8bf bhi = *(const v8bf*)(bp + 16);
      v16bf bfrag;
#pragma unroll
      for (int i = 0; i < 8; ++i) { bfrag[i] = blo[i]; bfrag[i + 8] = bhi[i]; }

      if (nt == 0) acc0 = __builtin_amdgcn_wmma_f32_16x16x32_bf16(false, afrag, false, bfrag, (short)0, acc0, false, false);
      if (nt == 1) acc1 = __builtin_amdgcn_wmma_f32_16x16x32_bf16(false, afrag, false, bfrag, (short)0, acc1, false, false);
      if (nt == 2) acc2 = __builtin_amdgcn_wmma_f32_16x16x32_bf16(false, afrag, false, bfrag, (short)0, acc2, false, false);
      if (nt == 3) acc3 = __builtin_amdgcn_wmma_f32_16x16x32_bf16(false, afrag, false, bfrag, (short)0, acc3, false, false);
    }
  }

  // C/D 16x16 f32 layout: VGPR r, lane-half hi -> row = r + 8*hi, col = lane&15
#pragma unroll
  for (int r = 0; r < 8; ++r) {
    float* out = H + (size_t)(mbase + r + 8 * hi) * DOUT + nbase + m;
    out[0]  = acc0[r];
    out[16] = acc1[r];
    out[32] = acc2[r];
    out[48] = acc3[r];
  }
}

// ---------------- attention logits per node/head ----------------
__global__ void alphas_kernel(const float* __restrict__ H,
                              const float* __restrict__ a_src, const float* __restrict__ a_dst,
                              float* __restrict__ AS, float* __restrict__ AD) {
  int idx = blockIdx.x * blockDim.x + threadIdx.x;    // over N_*HEADS
  int n = idx >> 2, h = idx & 3;
  const float* hp = H + (size_t)n * DOUT + h * HID;
  const float* a1 = a_src + h * HID;
  const float* a2 = a_dst + h * HID;
  float s1 = 0.f, s2 = 0.f;
#pragma unroll 8
  for (int k = 0; k < HID; ++k) { float hv = hp[k]; s1 += hv * a1[k]; s2 += hv * a2[k]; }
  AS[idx] = s1;
  AD[idx] = s2;
}

// ---------------- edge pass 1: leaky-relu logit + segment max ----------------
__global__ void edge_logits_max_kernel(const int* __restrict__ ei,
                                       const float* __restrict__ AS, const float* __restrict__ AD,
                                       float* __restrict__ ebuf, float* __restrict__ mbuf) {
  int idx = blockIdx.x * blockDim.x + threadIdx.x;    // over ET_*HEADS
  int e = idx >> 2, h = idx & 3;
  int s, d;
  if (e < E_) { s = ei[e]; d = ei[E_ + e]; } else { s = d = e - E_; }
  float v = AS[s * HEADS + h] + AD[d * HEADS + h];
  v = (v > 0.f) ? v : NSLOPE * v;
  ebuf[idx] = v;
  atomicMaxF32(&mbuf[d * HEADS + h], v);
}

// ---------------- edge pass 2: exp + segment sum ----------------
__global__ void edge_expsum_kernel(const int* __restrict__ ei,
                                   float* __restrict__ ebuf,
                                   const float* __restrict__ mbuf, float* __restrict__ den) {
  int idx = blockIdx.x * blockDim.x + threadIdx.x;
  int e = idx >> 2, h = idx & 3;
  int d;
  if (e < E_) { d = ei[E_ + e]; } else { d = e - E_; }
  float ex = __expf(ebuf[idx] - mbuf[d * HEADS + h]);
  ebuf[idx] = ex;
  atomicAdd(&den[d * HEADS + h], ex);
}

// ---------------- edge pass 3: weighted message scatter ----------------
__global__ void edge_scatter_kernel(const int* __restrict__ ei,
                                    const float* __restrict__ ebuf, const float* __restrict__ den,
                                    const float* __restrict__ H, float* __restrict__ AGG) {
  int idx = blockIdx.x * blockDim.x + threadIdx.x;    // over ET_*HEADS
  int e = idx >> 2, h = idx & 3;
  int s, d;
  if (e < E_) { s = ei[e]; d = ei[E_ + e]; } else { s = d = e - E_; }
  float alpha = ebuf[idx] / den[d * HEADS + h];
  const float4* hs = (const float4*)(H + (size_t)s * DOUT + h * HID);
  float* ag = AGG + (size_t)d * DOUT + h * HID;
#pragma unroll 4
  for (int k = 0; k < HID / 4; ++k) {
    float4 hv = hs[k];
    atomicAdd(ag + 4 * k + 0, hv.x * alpha);
    atomicAdd(ag + 4 * k + 1, hv.y * alpha);
    atomicAdd(ag + 4 * k + 2, hv.z * alpha);
    atomicAdd(ag + 4 * k + 3, hv.w * alpha);
  }
}

// ---------------- finalize layer: head mean + bias + relu, pool, next input ----------------
__global__ void finalize_kernel(const float* __restrict__ AGG, const float* __restrict__ bias,
                                float* __restrict__ pooled, __bf16* __restrict__ XbNext,
                                int layer) {
  int idx = blockIdx.x * blockDim.x + threadIdx.x;    // over N_*HID
  int n = idx >> 6, d = idx & 63;
  const float* ag = AGG + (size_t)n * DOUT;
  float v = (ag[d] + ag[HID + d] + ag[2 * HID + d] + ag[3 * HID + d]) * 0.25f + bias[d];
  v = fmaxf(v, 0.0f);
  int g = n >> 5;   // NODES_PER_G == 32
  atomicAdd(&pooled[((size_t)g * 3 + layer) * HID + d], v);
  XbNext[(size_t)n * KPAD + d] = (__bf16)v;
}

// ---------------- graph readout ----------------
__global__ void readout_kernel(const float* __restrict__ pooled,
                               const float* __restrict__ pw, const float* __restrict__ pb,
                               float* __restrict__ out) {
  int idx = blockIdx.x * blockDim.x + threadIdx.x;    // over G_*HID
  int g = idx >> 6, d = idx & 63;
  const float* pg = pooled + (size_t)g * 3 * HID;
  float logit[3];
#pragma unroll
  for (int l = 0; l < 3; ++l) {
    float s = 0.f;
#pragma unroll 8
    for (int k = 0; k < HID; ++k) s += pg[l * HID + k] * pw[k];
    logit[l] = s + pb[0];
  }
  float mx = fmaxf(logit[0], fmaxf(logit[1], logit[2]));
  float e0 = __expf(logit[0] - mx), e1 = __expf(logit[1] - mx), e2 = __expf(logit[2] - mx);
  float inv = 1.0f / (e0 + e1 + e2);
  out[idx] = (e0 * pg[d] + e1 * pg[HID + d] + e2 * pg[2 * HID + d]) * inv;
}

// ---------------- launcher ----------------
extern "C" void kernel_launch(void* const* d_in, const int* in_sizes, int n_in,
                              void* d_out, int out_size, void* d_ws, size_t ws_size,
                              hipStream_t stream) {
  (void)in_sizes; (void)n_in; (void)out_size; (void)ws_size;

  const float* x  = (const float*)d_in[0];
  const int*   ei = (const int*)d_in[1];
  const float* W[3]  = { (const float*)d_in[3],  (const float*)d_in[7],  (const float*)d_in[11] };
  const float* As[3] = { (const float*)d_in[4],  (const float*)d_in[8],  (const float*)d_in[12] };
  const float* Ad[3] = { (const float*)d_in[5],  (const float*)d_in[9],  (const float*)d_in[13] };
  const float* Bs[3] = { (const float*)d_in[6],  (const float*)d_in[10], (const float*)d_in[14] };
  const float* pw = (const float*)d_in[15];
  const float* pb = (const float*)d_in[16];
  float* out = (float*)d_out;

  // workspace carve (256B aligned)
  char* ws = (char*)d_ws;
  size_t off = 0;
  auto carve = [&](size_t bytes) -> char* {
    char* p = ws + off;
    off = (off + bytes + 255) & ~(size_t)255;
    return p;
  };
  float*  H    = (float*) carve((size_t)N_ * DOUT * 4);   // 134 MB
  float*  AGG  = (float*) carve((size_t)N_ * DOUT * 4);   // 134 MB
  __bf16* XB   = (__bf16*)carve((size_t)N_ * KPAD * 2);   // 25 MB
  __bf16* WT   = (__bf16*)carve((size_t)DOUT * KPAD * 2); // 48 KB
  float*  AS   = (float*) carve((size_t)N_ * HEADS * 4);
  float*  AD   = (float*) carve((size_t)N_ * HEADS * 4);
  float*  MB   = (float*) carve((size_t)N_ * HEADS * 4);
  float*  DEN  = (float*) carve((size_t)N_ * HEADS * 4);
  float*  EB   = (float*) carve((size_t)ET_ * HEADS * 4); // 10.5 MB
  float*  POOL = (float*) carve((size_t)G_ * 3 * HID * 4);

  const int T = 256;

  // prep inputs
  convert_x_kernel<<<(N_ * KPAD) / T, T, 0, stream>>>(x, XB);
  set_f32_kernel<<<3072, T, 0, stream>>>(POOL, 0.0f, (unsigned long long)G_ * 3 * HID);

  const int din[3]    = { INDIM, HID, HID };
  const int ksteps[3] = { 3, 2, 2 };

  for (int l = 0; l < 3; ++l) {
    prep_w_kernel<<<(DOUT * KPAD) / T, T, 0, stream>>>(W[l], WT, din[l]);
    gemm_wmma_kernel<<<N_ / 16, 128, 0, stream>>>(XB, WT, H, ksteps[l]);

    alphas_kernel<<<(N_ * HEADS) / T, T, 0, stream>>>(H, As[l], Ad[l], AS, AD);

    set_f32_kernel<<<2048, T, 0, stream>>>(MB, -INFINITY, (unsigned long long)N_ * HEADS);
    set_f32_kernel<<<2048, T, 0, stream>>>(DEN, 0.0f, (unsigned long long)N_ * HEADS);
    set_f32_kernel<<<8192, T, 0, stream>>>(AGG, 0.0f, (unsigned long long)N_ * DOUT);

    edge_logits_max_kernel<<<(ET_ * HEADS) / T, T, 0, stream>>>(ei, AS, AD, EB, MB);
    edge_expsum_kernel<<<(ET_ * HEADS) / T, T, 0, stream>>>(ei, EB, MB, DEN);
    edge_scatter_kernel<<<(ET_ * HEADS) / T, T, 0, stream>>>(ei, EB, DEN, H, AGG);

    finalize_kernel<<<(N_ * HID) / T, T, 0, stream>>>(AGG, Bs[l], POOL, XB, l);
  }

  readout_kernel<<<(G_ * HID) / T, T, 0, stream>>>(POOL, pw, pb, out);
}
